// ChunkGatedAttentionUnit_16535624090001
// MI455X (gfx1250) — matrix operand order repaired and checked by reading, not compiled, not run on hardware
//
#include <hip/hip_runtime.h>

typedef __bf16 bf16_t;
typedef bf16_t v16bf __attribute__((ext_vector_type(16)));
typedef bf16_t v8bf  __attribute__((ext_vector_type(8)));
typedef float  v8f   __attribute__((ext_vector_type(8)));
typedef unsigned int u32x4 __attribute__((ext_vector_type(4)));
typedef int i32x4 __attribute__((ext_vector_type(4)));
typedef int i32x8 __attribute__((ext_vector_type(8)));

#define DEV_INLINE __device__ __forceinline__

constexpr int  Bb  = 2;
constexpr int  S   = 4096;
constexpr int  H   = 1024;
constexpr int  D   = 2048;
constexpr int  CSZ = 128;
constexpr int  NC  = S / CSZ;          // 32 chunks per batch
constexpr long MR  = (long)Bb * S;     // 8192 total rows

// GEMM mode flags
constexpr int GM_ACCUM   = 1;   // out += existing f32 C
constexpr int GM_BIAS    = 2;   // add bias[n]
constexpr int GM_SIGMOID = 4;   // apply sigmoid
constexpr int GM_OUTBF16 = 8;   // write bf16 (else f32)
constexpr int GM_MIRROR  = 16;  // also write bf16 mirror copy

DEV_INLINE v8f wmma_bf16f32(v16bf a, v16bf b, v8f c) {
  // D = A(16x32,bf16) * B(32x16,bf16) + C(16x16,f32)
  return __builtin_amdgcn_wmma_f32_16x16x32_bf16(false, a, false, b,
                                                 (short)0, c, false, false);
}

// A fragment, row-major A[M][K]: lane l<16 -> row m0+l, K {k0..k0+7, k0+16..k0+23};
// lanes 16..31 -> same rows, K {k0+8..15, k0+24..31}  (ISA 7.12.2, 16-bit A 16x32)
DEV_INLINE v16bf load_frag_a(const bf16_t* A, int lda, int m0, int k0) {
  const int lane = threadIdx.x & 31;
  const int hh = lane >> 4;
  const int l  = lane & 15;
  const bf16_t* p = A + (size_t)(m0 + l) * (size_t)lda + k0 + hh * 8;
  v8bf lo = *(const v8bf*)(p);
  v8bf hi = *(const v8bf*)(p + 16);
  v16bf f;
#pragma unroll
  for (int i = 0; i < 8; ++i) { f[i] = lo[i]; f[i + 8] = hi[i]; }
  return f;
}

// B fragment, row-major B[K][N]: lane l -> col n0+(l&15), K {k0+8h..+7, k0+16+8h..+7}, h=l>>4
DEV_INLINE v16bf load_frag_b(const bf16_t* B, int ldb, int k0, int n0) {
  const int lane = threadIdx.x & 31;
  const bf16_t* p = B + (size_t)(k0 + (lane >> 4) * 8) * (size_t)ldb + n0 + (lane & 15);
  v16bf f;
#pragma unroll
  for (int i = 0; i < 8; ++i) { f[i] = p[(size_t)i * ldb]; f[i + 8] = p[(size_t)(i + 16) * ldb]; }
  return f;
}

// ---------------------------------------------------------------------------
// Tensor Data Mover: 2D tile (tile_w x tile_h elements, 2-byte elems) from a
// row-major tensor (row_stride elems) into LDS at lds_off. D# per ISA 08 §8.
// 6-arg builtin variant (clang-23 / therock-10.0 headers).
// ---------------------------------------------------------------------------
DEV_INLINE void tdm_load_2d_bf16(unsigned int lds_off, const void* gptr,
                                 unsigned int tile_w, unsigned int tile_h,
                                 unsigned int tensor_w, unsigned int tensor_h,
                                 unsigned long long row_stride) {
  const unsigned long long ga = (unsigned long long)gptr;
  u32x4 g0;
  g0[0] = 1u;                                   // count=1, user descriptor
  g0[1] = lds_off;                              // lds_addr (bytes)
  g0[2] = (unsigned int)ga;                     // global_addr[31:0]
  g0[3] = (unsigned int)((ga >> 32) & 0x01FFFFFFu) | (2u << 30);  // addr hi, type=2
  i32x8 g1;
  g1[0] = (int)(1u << 16);                      // data_size=1 (2 bytes), no mask
  g1[1] = (int)((tensor_w & 0xFFFFu) << 16);    // tensor_dim0[15:0]
  g1[2] = (int)((tensor_w >> 16) | ((tensor_h & 0xFFFFu) << 16));
  g1[3] = (int)((tensor_h >> 16) | (tile_w << 16));   // tile_dim0
  g1[4] = (int)(tile_h & 0xFFFFu);              // tile_dim1 (tile_dim2=0)
  g1[5] = (int)(unsigned int)(row_stride & 0xFFFFFFFFull);  // dim0_stride lo32
  g1[6] = (int)(unsigned int)((row_stride >> 32) & 0xFFFFull);
  g1[7] = 0;
  i32x4 zz4;
  zz4[0] = 0; zz4[1] = 0; zz4[2] = 0; zz4[3] = 0;
  i32x8 zz8;
#pragma unroll
  for (int i = 0; i < 8; ++i) zz8[i] = 0;
  __builtin_amdgcn_tensor_load_to_lds(g0, g1, zz4, zz4, zz8, 0);
}

// ---------------------------------------------------------------------------
// Generic batched bf16 WMMA GEMM, TDM-staged through LDS, double-buffered.
// Block = 256 threads = 8 waves; wave tile 32x64; block tile 64x256.
// Per 64-deep K-step wave0 TDM-loads A(64x64) and B(64x256) tiles (8+32 KB
// per buffer, 2 buffers = 80 KB LDS); waves run 16 WMMAs per barrier interval
// out of LDS.   grid = (N/256, M/64, batch); K must be a multiple of 64.
// ---------------------------------------------------------------------------
template <int MODE>
__global__ void __launch_bounds__(256)
gemm_bf16(const bf16_t* __restrict__ Ag, const bf16_t* __restrict__ Bg,
          const float* __restrict__ bias, float* __restrict__ Cf,
          bf16_t* __restrict__ Cb, bf16_t* __restrict__ Cm,
          int M, int N, int K, int lda, int ldb, int ldc,
          size_t sA, size_t sB, size_t sC) {
  __shared__ __align__(16) bf16_t shA[2][64 * 64];    // 8 KB per buffer
  __shared__ __align__(16) bf16_t shB[2][64 * 256];   // 32 KB per buffer

  const int z = blockIdx.z;
  Ag += sA * z;
  Bg += sB * z;
  const size_t coff = sC * z;

  const int lane = threadIdx.x & 31;
  const int wv   = threadIdx.x >> 5;
  const int hh   = lane >> 4;
  const int l    = lane & 15;
  const int bm = blockIdx.y * 64;
  const int bn = blockIdx.x * 256;
  const int m0w = (wv & 1) * 32;        // within block tile
  const int n0w = (wv >> 1) * 64;

  const unsigned int ldsA0 = (unsigned int)(size_t)(void*)&shA[0][0];
  const unsigned int ldsA1 = (unsigned int)(size_t)(void*)&shA[1][0];
  const unsigned int ldsB0 = (unsigned int)(size_t)(void*)&shB[0][0];
  const unsigned int ldsB1 = (unsigned int)(size_t)(void*)&shB[1][0];

  v8f acc[2][4];
#pragma unroll
  for (int mi = 0; mi < 2; ++mi)
#pragma unroll
    for (int j = 0; j < 4; ++j)
#pragma unroll
      for (int r = 0; r < 8; ++r) acc[mi][j][r] = 0.0f;

  // prologue: async-load tiles for k0 = 0 into buffer 0
  if (threadIdx.x < 32) {
    tdm_load_2d_bf16(ldsA0, Ag + (size_t)bm * lda, 64, 64,
                     (unsigned)K, (unsigned)M, (unsigned long long)lda);
    tdm_load_2d_bf16(ldsB0, Bg + bn, 256, 64,
                     (unsigned)N, (unsigned)K, (unsigned long long)ldb);
  }

  int buf = 0;
  for (int k0 = 0; k0 < K; k0 += 64) {
    const bool more = (k0 + 64 < K);
    if (threadIdx.x < 32) {
      if (more) {  // prefetch next K-step into the other buffer
        const unsigned int la = buf ? ldsA0 : ldsA1;
        const unsigned int lb = buf ? ldsB0 : ldsB1;
        tdm_load_2d_bf16(la, Ag + (size_t)bm * lda + (k0 + 64), 64, 64,
                         (unsigned)K, (unsigned)M, (unsigned long long)lda);
        tdm_load_2d_bf16(lb, Bg + (size_t)(k0 + 64) * ldb + bn, 256, 64,
                         (unsigned)N, (unsigned)K, (unsigned long long)ldb);
        __builtin_amdgcn_s_wait_tensorcnt(2);  // current 2 tiles have landed
      } else {
        __builtin_amdgcn_s_wait_tensorcnt(0);
      }
    }
    __syncthreads();  // tiles visible to all waves

#pragma unroll
    for (int kk = 0; kk < 64; kk += 32) {
      v16bf a0 = load_frag_a(shA[buf], 64, m0w, kk);
      v16bf a1 = load_frag_a(shA[buf], 64, m0w + 16, kk);
#pragma unroll
      for (int j = 0; j < 4; ++j) {
        v16bf b = load_frag_b(shB[buf], 256, kk, n0w + 16 * j);
        acc[0][j] = wmma_bf16f32(a0, b, acc[0][j]);
        acc[1][j] = wmma_bf16f32(a1, b, acc[1][j]);
      }
    }
    __syncthreads();  // everyone done reading before buffer is overwritten
    buf ^= 1;
  }

  // store
#pragma unroll
  for (int mi = 0; mi < 2; ++mi) {
#pragma unroll
    for (int j = 0; j < 4; ++j) {
      const int nn = bn + n0w + 16 * j + l;
      const float bv = (MODE & GM_BIAS) ? bias[nn] : 0.0f;
#pragma unroll
      for (int r = 0; r < 8; ++r) {
        const int mm = bm + m0w + 16 * mi + r + 8 * hh;
        const size_t idx = coff + (size_t)mm * (size_t)ldc + nn;
        float x = acc[mi][j][r] + bv;
        if (MODE & GM_ACCUM) x += Cf[idx];
        if (MODE & GM_SIGMOID) x = 1.0f / (1.0f + __expf(-x));
        if (MODE & GM_OUTBF16) Cb[idx] = (bf16_t)x;
        else                   Cf[idx] = x;
        if (MODE & GM_MIRROR)  Cm[idx] = (bf16_t)x;
      }
    }
  }
}

// ---------------------------------------------------------------------------
// Fused per-chunk causal softmax attention:
//   P = softmax(mask(Q Kt / sqrt(D))); local = P @ V  -> accF (f32)
// one block (8 waves) per (batch,chunk); wave w owns score rows 16w..16w+15.
// ---------------------------------------------------------------------------
__global__ void __launch_bounds__(256)
attn_local(const bf16_t* __restrict__ q, const bf16_t* __restrict__ kt,
           const bf16_t* __restrict__ v, float* __restrict__ accF) {
  constexpr int LDP = CSZ + 8;  // padded LDS stride (bank-conflict-free)
  __shared__ __align__(16) bf16_t Pl[CSZ * LDP];

  const int bc   = blockIdx.x;  // b*NC + c ; chunk rows = bc*CSZ .. +127
  const int lane = threadIdx.x & 31;
  const int wv   = threadIdx.x >> 5;
  const int hh   = lane >> 4;
  const int l    = lane & 15;
  const int m0   = wv * 16;

  const bf16_t* Qc  = q  + (size_t)bc * CSZ * D;
  const bf16_t* KTc = kt + (size_t)bc * D * CSZ;   // [D][CSZ]
  const bf16_t* Vc  = v  + (size_t)bc * CSZ * D;

  // ---- scores: 16 rows x 128 cols, contraction over D (pipelined) ----
  v8f s[8];
#pragma unroll
  for (int t = 0; t < 8; ++t)
#pragma unroll
    for (int r = 0; r < 8; ++r) s[t][r] = 0.0f;

  v16bf a = load_frag_a(Qc, D, m0, 0);
  v16bf bb[8];
#pragma unroll
  for (int t = 0; t < 8; ++t) bb[t] = load_frag_b(KTc, CSZ, 0, t * 16);

  for (int d0 = 32; d0 < D; d0 += 32) {
    // stream hint for the KT operand a few steps ahead
    __builtin_prefetch(KTc + (size_t)(d0 + 224 < D ? d0 + 224 : d0) * CSZ +
                           lane * CSZ, 0, 1);
    v16bf na = load_frag_a(Qc, D, m0, d0);
    v16bf nb[8];
#pragma unroll
    for (int t = 0; t < 8; ++t) nb[t] = load_frag_b(KTc, CSZ, d0, t * 16);
#pragma unroll
    for (int t = 0; t < 8; ++t) s[t] = wmma_bf16f32(a, bb[t], s[t]);
    a = na;
#pragma unroll
    for (int t = 0; t < 8; ++t) bb[t] = nb[t];
  }
#pragma unroll
  for (int t = 0; t < 8; ++t) s[t] = wmma_bf16f32(a, bb[t], s[t]);

  // ---- scale, causal mask, row softmax, write P to LDS as bf16 ----
  const float scale = 0.022097086912079612f;  // 1/sqrt(2048)
#pragma unroll
  for (int r = 0; r < 8; ++r) {
    const int row = m0 + r + 8 * hh;
    float vals[8];
    float mx = -1e30f;
#pragma unroll
    for (int t = 0; t < 8; ++t) {
      const int col = t * 16 + l;
      float x = s[t][r] * scale;
      if (col > row) x = -1e30f;
      vals[t] = x;
      mx = fmaxf(mx, x);
    }
#pragma unroll
    for (int off = 8; off >= 1; off >>= 1)
      mx = fmaxf(mx, __shfl_xor(mx, off, 32));
    float sum = 0.0f;
#pragma unroll
    for (int t = 0; t < 8; ++t) {
      float e = __expf(vals[t] - mx);
      vals[t] = e;
      sum += e;
    }
#pragma unroll
    for (int off = 8; off >= 1; off >>= 1)
      sum += __shfl_xor(sum, off, 32);
    const float inv = 1.0f / sum;
#pragma unroll
    for (int t = 0; t < 8; ++t)
      Pl[row * LDP + t * 16 + l] = (bf16_t)(vals[t] * inv);
  }
  __syncthreads();

  // ---- local = P @ V ; P A-frags hoisted from LDS, V B-frags grouped ----
  v16bf pa[4];
#pragma unroll
  for (int kk = 0; kk < 4; ++kk) pa[kk] = load_frag_a(Pl, LDP, m0, kk * 32);

  float* outc = accF + (size_t)bc * CSZ * D;
  for (int n0 = 0; n0 < D; n0 += 64) {
    v16bf vb2[4][4];
#pragma unroll
    for (int kk = 0; kk < 4; ++kk)
#pragma unroll
      for (int j = 0; j < 4; ++j)
        vb2[kk][j] = load_frag_b(Vc, D, kk * 32, n0 + 16 * j);

    v8f acc[4];
#pragma unroll
    for (int j = 0; j < 4; ++j)
#pragma unroll
      for (int r = 0; r < 8; ++r) acc[j][r] = 0.0f;
#pragma unroll
    for (int kk = 0; kk < 4; ++kk)
#pragma unroll
      for (int j = 0; j < 4; ++j)
        acc[j] = wmma_bf16f32(pa[kk], vb2[kk][j], acc[j]);

#pragma unroll
    for (int j = 0; j < 4; ++j)
#pragma unroll
      for (int r = 0; r < 8; ++r)
        outc[(size_t)(m0 + r + 8 * hh) * D + n0 + 16 * j + l] = acc[j][r];
  }
}

// ---------------------------------------------------------------------------
// Small helpers
// ---------------------------------------------------------------------------
__global__ void cvt_f32_bf16(const float* __restrict__ src,
                             bf16_t* __restrict__ dst, long n) {
  long i = (long)blockIdx.x * blockDim.x + threadIdx.x;
  const long stride = (long)gridDim.x * blockDim.x;
  for (; i < n; i += stride) dst[i] = (bf16_t)src[i];
}

__global__ void zero_f32(float* __restrict__ p, long n) {
  long i = (long)blockIdx.x * blockDim.x + threadIdx.x;
  const long stride = (long)gridDim.x * blockDim.x;
  for (; i < n; i += stride) p[i] = 0.0f;
}

// y = bf16(acc * gate)   (gate already sigmoid'ed, bf16)
__global__ void combine_gate(const float* __restrict__ acc,
                             const bf16_t* __restrict__ g,
                             bf16_t* __restrict__ y, long n) {
  long i = (long)blockIdx.x * blockDim.x + threadIdx.x;
  const long stride = (long)gridDim.x * blockDim.x;
  for (; i < n; i += stride) y[i] = (bf16_t)(acc[i] * (float)g[i]);
}

// per-chunk transpose: kT[bc][d][t] = k[bc*CSZ + t][d]; 32x32 LDS tiles
__global__ void __launch_bounds__(256)
transpose_chunks(const bf16_t* __restrict__ kk, bf16_t* __restrict__ kt) {
  __shared__ bf16_t tile[32][33];
  const int bc = blockIdx.z;
  const int d0 = blockIdx.x * 32;
  const int t0 = blockIdx.y * 32;
  const bf16_t* kc = kk + (size_t)bc * CSZ * D;
#pragma unroll
  for (int i = 0; i < 4; ++i) {
    const int t = t0 + threadIdx.y + 8 * i;
    tile[threadIdx.y + 8 * i][threadIdx.x] =
        kc[(size_t)t * D + d0 + threadIdx.x];
  }
  __syncthreads();
  bf16_t* ktc = kt + (size_t)bc * D * CSZ;
#pragma unroll
  for (int i = 0; i < 4; ++i) {
    const int d = d0 + threadIdx.y + 8 * i;
    ktc[(size_t)d * CSZ + t0 + threadIdx.x] = tile[threadIdx.x][threadIdx.y + 8 * i];
  }
}

// ---------------------------------------------------------------------------
// Host orchestration
// ---------------------------------------------------------------------------
extern "C" void kernel_launch(void* const* d_in, const int* in_sizes, int n_in,
                              void* d_out, int out_size, void* d_ws,
                              size_t ws_size, hipStream_t stream) {
  (void)in_sizes; (void)n_in; (void)out_size; (void)ws_size;

  const float* hs = (const float*)d_in[0];
  const float* Wq = (const float*)d_in[1];
  const float* bq = (const float*)d_in[2];
  const float* Wk = (const float*)d_in[3];
  const float* bk = (const float*)d_in[4];
  const float* Wv = (const float*)d_in[5];
  const float* bv = (const float*)d_in[6];
  const float* Wg = (const float*)d_in[7];
  const float* bg = (const float*)d_in[8];
  const float* Wo = (const float*)d_in[9];
  const float* bo = (const float*)d_in[10];

  char* ws = (char*)d_ws;
  size_t off = 0;
  auto alloc = [&](size_t bytes) -> void* {
    void* p = ws + off;
    off += (bytes + 255) & ~(size_t)255;
    return p;
  };

  bf16_t* hsb = (bf16_t*)alloc((size_t)MR * H * 2);
  bf16_t* wqb = (bf16_t*)alloc((size_t)H * D * 2);
  bf16_t* wkb = (bf16_t*)alloc((size_t)H * D * 2);
  bf16_t* wvb = (bf16_t*)alloc((size_t)H * D * 2);
  bf16_t* wgb = (bf16_t*)alloc((size_t)H * D * 2);
  bf16_t* wob = (bf16_t*)alloc((size_t)D * H * 2);
  bf16_t* qb  = (bf16_t*)alloc((size_t)MR * D * 2);
  bf16_t* kb  = (bf16_t*)alloc((size_t)MR * D * 2);
  bf16_t* vb  = (bf16_t*)alloc((size_t)MR * D * 2);
  bf16_t* gb  = (bf16_t*)alloc((size_t)MR * D * 2);
  bf16_t* ktb = (bf16_t*)alloc((size_t)MR * D * 2);
  float*  accF = (float*)alloc((size_t)MR * D * 4);
  float*  stF  = (float*)alloc((size_t)Bb * D * D * 4);
  bf16_t* stB  = (bf16_t*)alloc((size_t)Bb * D * D * 2);
  bf16_t* yb   = (bf16_t*)alloc((size_t)MR * D * 2);

  const dim3 blk(256);

  // 1) f32 -> bf16 conversions
  cvt_f32_bf16<<<2048, blk, 0, stream>>>(hs, hsb, MR * (long)H);
  cvt_f32_bf16<<<1024, blk, 0, stream>>>(Wq, wqb, (long)H * D);
  cvt_f32_bf16<<<1024, blk, 0, stream>>>(Wk, wkb, (long)H * D);
  cvt_f32_bf16<<<1024, blk, 0, stream>>>(Wv, wvb, (long)H * D);
  cvt_f32_bf16<<<1024, blk, 0, stream>>>(Wg, wgb, (long)H * D);
  cvt_f32_bf16<<<1024, blk, 0, stream>>>(Wo, wob, (long)D * H);

  // 2) projections: [MR,H] @ [H,D] (+bias) -> bf16   (gate: + sigmoid)
  {
    dim3 g(D / 256, MR / 64, 1);
    gemm_bf16<GM_BIAS | GM_OUTBF16><<<g, blk, 0, stream>>>(
        hsb, wqb, bq, nullptr, qb, nullptr, (int)MR, D, H, H, D, D, 0, 0, 0);
    gemm_bf16<GM_BIAS | GM_OUTBF16><<<g, blk, 0, stream>>>(
        hsb, wkb, bk, nullptr, kb, nullptr, (int)MR, D, H, H, D, D, 0, 0, 0);
    gemm_bf16<GM_BIAS | GM_OUTBF16><<<g, blk, 0, stream>>>(
        hsb, wvb, bv, nullptr, vb, nullptr, (int)MR, D, H, H, D, D, 0, 0, 0);
    gemm_bf16<GM_BIAS | GM_SIGMOID | GM_OUTBF16><<<g, blk, 0, stream>>>(
        hsb, wgb, bg, nullptr, gb, nullptr, (int)MR, D, H, H, D, D, 0, 0, 0);
  }

  // 3) per-chunk k transpose for contiguous WMMA B/A fragments
  {
    dim3 g(D / 32, CSZ / 32, Bb * NC);
    dim3 b(32, 8);
    transpose_chunks<<<g, b, 0, stream>>>(kb, ktb);
  }

  // 4) intra-chunk causal softmax attention -> accF (local term)
  attn_local<<<Bb * NC, blk, 0, stream>>>(qb, ktb, vb, accF);

  // 5) inter-chunk linear-attention scan (sequential over chunks,
  //    batched over B via grid.z; stream ordering gives the dependency)
  zero_f32<<<2048, blk, 0, stream>>>(stF, (long)Bb * D * D);
  for (int i = 0; i < NC; ++i) {
    if (i > 0) {
      // accF[chunk i] += Q_i @ state     (M=128, N=D, K=D)
      dim3 g(D / 256, CSZ / 64, Bb);
      gemm_bf16<GM_ACCUM><<<g, blk, 0, stream>>>(
          qb + (size_t)i * CSZ * D, stB, nullptr,
          accF + (size_t)i * CSZ * D, nullptr, nullptr,
          CSZ, D, D, D, D, D,
          (size_t)S * D, (size_t)D * D, (size_t)S * D);
    }
    if (i < NC - 1) {
      // state += Kt_i @ V_i   (M=D, N=D, K=128); also refresh bf16 mirror
      dim3 g(D / 256, D / 64, Bb);
      gemm_bf16<GM_ACCUM | GM_MIRROR><<<g, blk, 0, stream>>>(
          ktb + (size_t)i * D * CSZ, vb + (size_t)i * CSZ * D, nullptr,
          stF, nullptr, stB,
          D, D, CSZ, CSZ, D, D,
          (size_t)S * D, (size_t)S * D, (size_t)D * D);
    }
  }

  // 6) gating: y = bf16((local + cross) * gate)
  combine_gate<<<4096, blk, 0, stream>>>(accF, gb, yb, MR * (long)D);

  // 7) output projection: [MR,D] @ [D,H] + bo -> f32 d_out
  {
    dim3 g(H / 256, MR / 64, 1);
    gemm_bf16<GM_BIAS><<<g, blk, 0, stream>>>(
        yb, wob, bo, (float*)d_out, nullptr, nullptr,
        (int)MR, H, D, D, H, H, 0, 0, 0);
  }
}
